// UniformityLoss_841813590664
// MI455X (gfx1250) — matrix operand order repaired and checked by reading, not compiled, not run on hardware
//
#include <hip/hip_runtime.h>
#include <math.h>

// Problem constants (from reference): features [32,256,128,128] f32,
// 8x8 avg-pool -> X[N=8192, D=256], L2-normalized rows, RBF_T = 2.
#define NB    32
#define CC    256
#define HH    128
#define WW    128
#define NN    8192          // 32 * 16 * 16
#define DD    256
#define TILE  128           // super-tile edge (8x8 WMMA tiles per block)
#define LDA   264           // padded LDS row stride (bf16 elems): 132 dwords -> bank stride 4
#define NSB   (NN / TILE)   // 64 super-blocks per edge
#define NBLK  (NSB * (NSB + 1) / 2)   // 2080 upper-triangular super-blocks

typedef __bf16 bf16_t;
typedef bf16_t v16bf __attribute__((ext_vector_type(16)));
typedef float  v8f   __attribute__((ext_vector_type(8)));

union Frag16 { uint4 q[2]; v16bf v; };

__device__ __forceinline__ unsigned short f2bf(float f) {
    // round-to-nearest-even f32 -> bf16
    unsigned int u = __float_as_uint(f);
    unsigned int r = (u + 0x7FFFu + ((u >> 16) & 1u)) >> 16;
    return (unsigned short)r;
}

// Map linear index p in [0, NBLK) to (bi, bj) with bi <= bj.
// Row i starts at start(i) = i*(2*NSB+1-i)/2 and has (NSB-i) entries.
__device__ __forceinline__ void tri_map(int p, int& bi, int& bj) {
    const double a = 2.0 * NSB + 1.0;                  // 129
    int i = (int)((a - sqrt(a * a - 8.0 * (double)p)) * 0.5);
    if (i < 0) i = 0;
    if (i > NSB - 1) i = NSB - 1;
    while (i + 1 <= NSB - 1 && ((i + 1) * (2 * NSB + 1 - (i + 1))) / 2 <= p) ++i;
    while ((i * (2 * NSB + 1 - i)) / 2 > p) --i;
    bi = i;
    bj = i + (p - (i * (2 * NSB + 1 - i)) / 2);
}

// ---------------------------------------------------------------------------
// Kernel 1: 8x8 average pool + row L2-normalize, emit bf16 X[n][c].
// One block per (b, yh) strip: thread c reads 8 full rows of 128 floats
// (every byte of each 128B line is consumed -> bandwidth-efficient),
// producing 16 pooled values (yw = 0..15). Block-wide reduction gives the
// 16 row norms; writes X rows coalesced (256 threads x contiguous bf16).
// ---------------------------------------------------------------------------
__global__ __launch_bounds__(256)
void pool_normalize_kernel(const float* __restrict__ feat,
                           unsigned short* __restrict__ xbf) {
    const int bid  = blockIdx.x;          // b*16 + yh
    const int c    = threadIdx.x;         // channel, 0..255
    const int lane = c & 31, wid = c >> 5;

    const long long base =
        ((long long)(bid >> 4) * CC + c) * (long long)(HH * WW) +
        (long long)(bid & 15) * 8 * WW;

    float acc[16];
#pragma unroll
    for (int yw = 0; yw < 16; ++yw) acc[yw] = 0.0f;

#pragma unroll
    for (int i = 0; i < 8; ++i) {
        const float4* row = reinterpret_cast<const float4*>(feat + base + (long long)i * WW);
#pragma unroll
        for (int yw = 0; yw < 16; ++yw) {
            float4 a = row[2 * yw];
            float4 b = row[2 * yw + 1];
            acc[yw] += (a.x + a.y + a.z + a.w) + (b.x + b.y + b.z + b.w);
        }
    }

    __shared__ float ssum[8][16];
    __shared__ float sscale[16];

#pragma unroll
    for (int yw = 0; yw < 16; ++yw) {
        acc[yw] *= (1.0f / 64.0f);              // pooled value
        float p = acc[yw] * acc[yw];
        for (int off = 16; off > 0; off >>= 1)  // wave32 reduction
            p += __shfl_down(p, off, 32);
        if (lane == 0) ssum[wid][yw] = p;
    }
    __syncthreads();

    if (c < 16) {
        float n2 = 0.0f;
#pragma unroll
        for (int w = 0; w < 8; ++w) n2 += ssum[w][c];
        sscale[c] = 1.0f / fmaxf(sqrtf(n2), 1e-12f);
    }
    __syncthreads();

#pragma unroll
    for (int yw = 0; yw < 16; ++yw) {
        const int n = bid * 16 + yw;            // n = b*256 + yh*16 + yw
        xbf[(size_t)n * DD + c] = f2bf(acc[yw] * sscale[yw]);
    }
}

// ---------------------------------------------------------------------------
// Kernel 2: symmetric Gram super-tile + fused RBF-sum epilogue.
// Only upper-triangular super-blocks (bi <= bj); off-diagonal partials are
// weighted x2. Each block: 128x128 output tile, A/B rows staged in padded
// LDS (132 KB dynamic). 8 waves in a 4x2 arrangement; each wave owns a
// 2x4 block of 16x16 tiles: per K-step 2 A frags + 4 B frags (12
// ds_load_b128) feed 8 v_wmma_f32_16x16x32_bf16 (vs 18 loads for 1x8).
// Epilogue: exp(-2*max(2-2g,0)) summed, wave-reduced, one partial per
// block (no atomics -> deterministic).
// ---------------------------------------------------------------------------
__global__ __launch_bounds__(256)
void gram_rbf_kernel(const unsigned short* __restrict__ xbf,
                     float* __restrict__ partials) {
    extern __shared__ unsigned short smem[];
    unsigned short* As = smem;                // [128][LDA]
    unsigned short* Bs = smem + TILE * LDA;   // [128][LDA]

    const int tid = threadIdx.x;
    int bi, bj;
    tri_map((int)blockIdx.x, bi, bj);
    const int i0 = bi * TILE;
    const int j0 = bj * TILE;

    // Stage global -> LDS: each thread copies half a row (256B) of A and B.
    {
        const int r = tid >> 1, half = (tid & 1) * 128;
        const uint4* gA = reinterpret_cast<const uint4*>(xbf + (size_t)(i0 + r) * DD + half);
        const uint4* gB = reinterpret_cast<const uint4*>(xbf + (size_t)(j0 + r) * DD + half);
        uint4* lA = reinterpret_cast<uint4*>(As + r * LDA + half);
        uint4* lB = reinterpret_cast<uint4*>(Bs + r * LDA + half);
#pragma unroll
        for (int q = 0; q < 16; ++q) { lA[q] = gA[q]; lB[q] = gB[q]; }
    }
    __syncthreads();

    const int wid = tid >> 5, lane = tid & 31;
    const int m = lane & 15, hi = lane >> 4;  // row-in-tile / K-half select
    const int wr = wid >> 1;                  // wave tile-row block (0..3) -> rows wr*2, wr*2+1
    const int wc = wid & 1;                   // wave tile-col block (0..1) -> cols wc*4 .. +3

    const v8f vzero = {0.f, 0.f, 0.f, 0.f, 0.f, 0.f, 0.f, 0.f};
    v8f acc[2][4];
#pragma unroll
    for (int a = 0; a < 2; ++a)
#pragma unroll
        for (int b = 0; b < 4; ++b) acc[a][b] = vzero;

    const unsigned short* aRow0 = As + ((wr * 2 + 0) * 16 + m) * LDA;
    const unsigned short* aRow1 = As + ((wr * 2 + 1) * 16 + m) * LDA;

#pragma unroll 1
    for (int kk = 0; kk < 8; ++kk) {          // K = kk*32 .. +31
        // A 16x32 bf16 fragments: lanes 0-15 hold K = hi*8..+7 (elems 0-7)
        // and K = 16+hi*8..+7 (elems 8-15); lanes 16-31 shifted by 8.
        Frag16 fa[2];
        fa[0].q[0] = *reinterpret_cast<const uint4*>(aRow0 + kk * 32 + hi * 8);
        fa[0].q[1] = *reinterpret_cast<const uint4*>(aRow0 + kk * 32 + 16 + hi * 8);
        fa[1].q[0] = *reinterpret_cast<const uint4*>(aRow1 + kk * 32 + hi * 8);
        fa[1].q[1] = *reinterpret_cast<const uint4*>(aRow1 + kk * 32 + 16 + hi * 8);

        // B 32x16 bf16 fragments: lane = column N (= row j0+jt*16+m of X),
        // same K layout as A: K = hi*8..+7 (elems 0-7), K = 16+hi*8..+7 (elems 8-15).
        Frag16 fb[4];
#pragma unroll
        for (int b = 0; b < 4; ++b) {
            const uint4* pb = reinterpret_cast<const uint4*>(
                Bs + ((wc * 4 + b) * 16 + m) * LDA + kk * 32 + hi * 8);
            fb[b].q[0] = pb[0];
            fb[b].q[1] = pb[2];
        }

#pragma unroll
        for (int a = 0; a < 2; ++a)
#pragma unroll
            for (int b = 0; b < 4; ++b)
                acc[a][b] = __builtin_amdgcn_wmma_f32_16x16x32_bf16(
                    false, fa[a].v, false, fb[b].v, (short)0, acc[a][b], false, false);
    }

    // Fused epilogue: rows normalized -> d2 = max(2 - 2g, 0); val = exp(-2*d2).
    float s = 0.0f;
#pragma unroll
    for (int a = 0; a < 2; ++a) {
#pragma unroll
        for (int b = 0; b < 4; ++b) {
#pragma unroll
            for (int e = 0; e < 8; ++e) {
                float g  = acc[a][b][e];
                float d2 = fmaxf(2.0f - 2.0f * g, 0.0f);
                s += __expf(-2.0f * d2);
            }
        }
    }
    for (int off = 16; off > 0; off >>= 1)
        s += __shfl_down(s, off, 32);

    __shared__ float wsum[8];
    if (lane == 0) wsum[wid] = s;
    __syncthreads();
    if (tid == 0) {
        float t = 0.0f;
#pragma unroll
        for (int w = 0; w < 8; ++w) t += wsum[w];
        // symmetric matrix: off-diagonal super-blocks counted twice
        partials[blockIdx.x] = (bi == bj) ? t : 2.0f * t;
    }
}

// ---------------------------------------------------------------------------
// Kernel 3: deterministic f64 reduction of NBLK partials + finalize.
// total equals the full NxN sum (including the diagonal, exactly 1.0 per
// element after the clamp), so subtract N, divide by N*(N-1), log.
// ---------------------------------------------------------------------------
__global__ __launch_bounds__(256)
void finalize_kernel(const float* __restrict__ partials, float* __restrict__ out) {
    __shared__ double sd[256];
    const int t = threadIdx.x;
    double local = 0.0;
    for (int idx = t; idx < NBLK; idx += 256)   // fixed order -> deterministic
        local += (double)partials[idx];
    sd[t] = local;
    __syncthreads();
    for (int sstep = 128; sstep > 0; sstep >>= 1) {
        if (t < sstep) sd[t] += sd[t + sstep];
        __syncthreads();
    }
    if (t == 0) {
        double total = sd[0];
        double mean  = (total - (double)NN) / ((double)NN * (double)(NN - 1));
        out[0] = (float)log(mean);
    }
}

// ---------------------------------------------------------------------------
extern "C" void kernel_launch(void* const* d_in, const int* in_sizes, int n_in,
                              void* d_out, int out_size, void* d_ws, size_t ws_size,
                              hipStream_t stream) {
    (void)in_sizes; (void)n_in; (void)out_size; (void)ws_size;

    const float* feat = (const float*)d_in[0];
    // workspace layout: [0, 4MB) bf16 X[N][D]; then NBLK float partials
    unsigned short* xbf = (unsigned short*)d_ws;
    float* partials = (float*)((char*)d_ws + (size_t)NN * DD * sizeof(unsigned short));
    float* out = (float*)d_out;

    pool_normalize_kernel<<<NB * 16, 256, 0, stream>>>(feat, xbf);

    const size_t smem_bytes = (size_t)2 * TILE * LDA * sizeof(unsigned short); // 132 KB
    gram_rbf_kernel<<<NBLK, 256, smem_bytes, stream>>>(xbf, partials);

    finalize_kernel<<<1, 256, 0, stream>>>(partials, out);
}
